// LSTM_70_42537356100406
// MI455X (gfx1250) — matrix operand order repaired and checked by reading, not compiled, not run on hardware
//
#include <hip/hip_runtime.h>
#include <math.h>

// ---------------------------------------------------------------------------
// 2-layer chained LSTM (T=70, B=256, I=H=512), fp32 in/out, per-step weights.
//
// MI455X reasoning:
//  * Weights are used exactly once: 70 * 4 * 2048*512*4B = 1.12 GB @ 23.3 TB/s
//    => ~48 us HBM floor. Compute is only 150 GFLOP, so stream fp32 weights
//    once and convert to bf16 in-register (pre-converting in memory would add
//    traffic). The h->h recurrence is sequential, so everything runs in ONE
//    persistent cooperative kernel with a device-wide barrier 2x per step.
//  * GEMMs use v_wmma_f32_16x16x32_bf16 (fp32 accumulate, bias preloaded in
//    the accumulator). WG tile: 64 batch rows x 32 hidden cols x 4 gates, so
//    the pointwise LSTM cell is WG-local (gates combined through LDS).
//  * Fragment layouts per CDNA5 ISA: A [B][K] row-major and
//    W [4H][K] row-major stage directly (no transposes), fragments are built
//    with two 16B LDS vector loads each.
//  * h0/h1 double-buffered by timestep parity in d_ws; c0/c1 owned per-element
//    by a fixed thread (no cross-WG sync needed for c).
// ---------------------------------------------------------------------------

#define TT   70
#define BB   256
#define HH   512
#define G4   2048              // 4*H
#define NWG  64                // 4 batch blocks x 16 hidden blocks
#define TM   64                // batch rows per WG
#define TN   32                // hidden cols per WG (x 4 gates)
#define KC   32                // K chunk per WMMA

typedef __attribute__((ext_vector_type(16))) __bf16 v16bf;
typedef __attribute__((ext_vector_type(8)))  __bf16 v8bf;
typedef __attribute__((ext_vector_type(4)))  __bf16 v4bf;
typedef __attribute__((ext_vector_type(8)))  float  v8f;
typedef __attribute__((ext_vector_type(4)))  float  v4f;

// workspace layout (floats)
#define BHf      (BB*HH)
#define OFF_BAR  0                      // [0]=count, [1]=phase
#define OFF_H0   64
#define OFF_C0   (OFF_H0 + 2*BHf)
#define OFF_H1   (OFF_C0 + BHf)
#define OFF_C1   (OFF_H1 + 2*BHf)
#define WS_TOT   (OFF_C1 + BHf)         // 786496 floats ~= 3.0 MB

__device__ __forceinline__ float sigm(float x) { return 1.0f / (1.0f + __expf(-x)); }

__device__ __forceinline__ v4bf cvt4(v4f v) {
    v4bf r;
    r[0] = (__bf16)v[0]; r[1] = (__bf16)v[1];
    r[2] = (__bf16)v[2]; r[3] = (__bf16)v[3];
    return r;
}

// device-wide split barrier: arrive (atomic add) + spin on monotonic phase
__device__ __forceinline__ void grid_barrier(unsigned* cnt, unsigned* ph, unsigned phase) {
    __syncthreads();
    if (threadIdx.x == 0) {
        __threadfence();
        unsigned old = __atomic_fetch_add(cnt, 1u, __ATOMIC_ACQ_REL);
        if (old == NWG - 1u) {
            __atomic_store_n(cnt, 0u, __ATOMIC_RELAXED);
            __atomic_store_n(ph, phase, __ATOMIC_RELEASE);
        } else {
            while (__atomic_load_n(ph, __ATOMIC_ACQUIRE) < phase) {
                __builtin_amdgcn_s_sleep(1);
            }
        }
    }
    __syncthreads();
}

// One LSTM layer-step for this WG's (64 x 32) slab:
//   g = bias + xin@Wih^T + hprev@Whh^T ; cell update ; write hnew (+ optional y)
__device__ __forceinline__ void lstm_layer(
    const float* __restrict__ xin,    // [256 x 512] row-major
    const float* __restrict__ Wih,    // [2048 x 512] row-major
    const float* __restrict__ hprev,  // [256 x 512]
    const float* __restrict__ Whh,    // [2048 x 512]
    const float* __restrict__ bi,     // [2048]
    const float* __restrict__ bh,     // [2048]
    float*       __restrict__ cst,    // [256 x 512]  (element-owned, in/out)
    float*       __restrict__ hnew,   // [256 x 512]
    float*       __restrict__ yout,   // optional [256 x 512]
    __bf16* As, __bf16* Bs, float (*Gs)[TM][TN],
    unsigned wgB0, unsigned wgN0)
{
    const unsigned tid  = threadIdx.x;
    const unsigned w    = tid >> 5;        // wave 0..7
    const unsigned lane = tid & 31;
    const unsigned mi   = w & 3;           // 16-row tile within 64 rows
    const unsigned gb   = (w >> 2) * 2;    // gates {0,1} or {2,3}
    const unsigned ln15 = lane & 15;
    const unsigned hi   = lane >> 4;       // lane-group select

    // accumulators [gate][ncol-tile], preloaded with bias (bias depends on n only,
    // and every element of a lane's v8f shares the same n -> broadcast is exact)
    v8f acc[2][2];
#pragma unroll
    for (int gg = 0; gg < 2; ++gg)
#pragma unroll
        for (int ni = 0; ni < 2; ++ni) {
            unsigned gidx = (gb + gg) * HH + wgN0 + ni * 16 + ln15;
            float bv = bi[gidx] + bh[gidx];
#pragma unroll
            for (int r = 0; r < 8; ++r) acc[gg][ni][r] = bv;
        }

    // two K=512 passes: xin@Wih^T then hprev@Whh^T
    for (int src = 0; src < 2; ++src) {
        const float* __restrict__ A = src ? hprev : xin;
        const float* __restrict__ W = src ? Whh : Wih;
        for (int k0 = 0; k0 < 512; k0 += KC) {
            // --- stage A: 64 rows x 32 K, fp32 -> bf16 (512 float4, 2/thread) ---
#pragma unroll
            for (int j = 0; j < 2; ++j) {
                unsigned idx = tid + 256u * j;              // 0..511
                unsigned r = idx >> 3, c4 = (idx & 7) * 4;
                v4f v = *(const v4f*)(A + (size_t)(wgB0 + r) * 512 + k0 + c4);
                *(v4bf*)(As + r * KC + c4) = cvt4(v);
            }
            // --- stage B: 4 gates x 32 rows x 32 K (1024 float4, 4/thread) ---
#pragma unroll
            for (int j = 0; j < 4; ++j) {
                unsigned idx = tid + 256u * j;              // 0..1023
                unsigned r = idx >> 3, c4 = (idx & 7) * 4;  // r: 0..127
                unsigned grow = (r >> 5) * HH + wgN0 + (r & 31);
                const float* wp = W + (size_t)grow * 512 + k0 + c4;
                v4f v = *(const v4f*)wp;
                *(v4bf*)(Bs + r * KC + c4) = cvt4(v);
                if (j == 0 && k0 + KC < 512) {
                    // pull next K-chunk of this weight row toward the caches
                    __builtin_prefetch(wp + KC, 0, 1);      // global_prefetch_b8
                }
            }
            __syncthreads();

            // --- A fragment (16x32 bf16): lane ln15 = M, hi selects K subsets
            //     K = {8*hi+0..7} then {16+8*hi+0..7} (per ISA 16-bit A table)
            unsigned arow = mi * 16 + ln15;
            v8bf a0 = *(const v8bf*)(As + arow * KC + 8 * hi);
            v8bf a1 = *(const v8bf*)(As + arow * KC + 16 + 8 * hi);
            v16bf afrag = __builtin_shufflevector(a0, a1,
                0, 1, 2, 3, 4, 5, 6, 7, 8, 9, 10, 11, 12, 13, 14, 15);

#pragma unroll
            for (int gg = 0; gg < 2; ++gg) {
#pragma unroll
                for (int ni = 0; ni < 2; ++ni) {
                    // B fragment (32x16 bf16): lane ln15 = N, K = {8*hi+0..7} then {16+8*hi+0..7}
                    unsigned brow = (gb + gg) * 32 + ni * 16 + ln15;
                    v8bf b0 = *(const v8bf*)(Bs + brow * KC + 8 * hi);
                    v8bf b1 = *(const v8bf*)(Bs + brow * KC + 16 + 8 * hi);
                    v16bf bfrag = __builtin_shufflevector(b0, b1,
                        0, 1, 2, 3, 4, 5, 6, 7, 8, 9, 10, 11, 12, 13, 14, 15);
                    acc[gg][ni] = __builtin_amdgcn_wmma_f32_16x16x32_bf16(
                        false, afrag, false, bfrag, (short)0, acc[gg][ni],
                        false, false);
                }
            }
            __syncthreads();
        }
    }

    // --- gates -> LDS (D layout: lane ln15 = N, vgpr r -> M = r + 8*hi) ---
#pragma unroll
    for (int gg = 0; gg < 2; ++gg)
#pragma unroll
        for (int ni = 0; ni < 2; ++ni)
#pragma unroll
            for (int r = 0; r < 8; ++r)
                Gs[gb + gg][mi * 16 + r + 8 * hi][ni * 16 + ln15] = acc[gg][ni][r];
    __syncthreads();

    // --- pointwise LSTM cell: 64*32 = 2048 elems, 8 per thread (fixed owner) ---
#pragma unroll
    for (int e = 0; e < 8; ++e) {
        unsigned lin = tid + 256u * e;
        unsigned row = lin >> 5, col = lin & 31;
        size_t gidx = (size_t)(wgB0 + row) * HH + wgN0 + col;
        float gi = Gs[0][row][col];
        float gf = Gs[1][row][col];
        float gc = Gs[2][row][col];
        float go = Gs[3][row][col];
        float c  = cst[gidx];
        c = sigm(gf) * c + sigm(gi) * tanhf(gc);
        float h = sigm(go) * tanhf(c);
        cst[gidx]  = c;
        hnew[gidx] = h;
        if (yout) yout[gidx] = h;
    }
    // (Gs is only rewritten after the next layer's full K loop, which contains
    //  multiple __syncthreads and a grid barrier -> no extra sync needed here)
}

__global__ void __launch_bounds__(256) lstm70_init_ws(float* __restrict__ ws) {
    size_t i = (size_t)blockIdx.x * blockDim.x + threadIdx.x;
    size_t stride = (size_t)gridDim.x * blockDim.x;
    for (size_t j = i; j < (size_t)WS_TOT; j += stride) ws[j] = 0.0f;
}

__global__ void __launch_bounds__(256) lstm70_persistent(
    const float* __restrict__ x,
    const float* __restrict__ Wih0, const float* __restrict__ Whh0,
    const float* __restrict__ bih0, const float* __restrict__ bhh0,
    const float* __restrict__ Wih1, const float* __restrict__ Whh1,
    const float* __restrict__ bih1, const float* __restrict__ bhh1,
    float* __restrict__ out, float* __restrict__ ws)
{
    __shared__ __bf16 As[TM * KC];        // 4 KB
    __shared__ __bf16 Bs[4 * TN * KC];    // 8 KB
    __shared__ float  Gs[4][TM][TN];      // 32 KB

    const unsigned wg   = blockIdx.x;
    const unsigned wgB0 = (wg >> 4) * TM; // 4 batch blocks of 64
    const unsigned wgN0 = (wg & 15) * TN; // 16 hidden blocks of 32

    unsigned* bar_cnt = (unsigned*)(ws + OFF_BAR);
    unsigned* bar_ph  = (unsigned*)(ws + OFF_BAR) + 1;
    float* h0ws = ws + OFF_H0;
    float* c0ws = ws + OFF_C0;
    float* h1ws = ws + OFF_H1;
    float* c1ws = ws + OFF_C1;

    unsigned phase = 0;
    unsigned p = 0; // h double-buffer parity
    for (int t = 0; t < TT; ++t) {
        // layer 0: consumes x_t and h0[p], produces h0[p^1], c0
        lstm_layer(x    + (size_t)t * BB * HH,
                   Wih0 + (size_t)t * G4 * HH,
                   h0ws + (size_t)p * BHf,
                   Whh0 + (size_t)t * G4 * HH,
                   bih0 + (size_t)t * G4, bhh0 + (size_t)t * G4,
                   c0ws, h0ws + (size_t)(p ^ 1) * BHf, nullptr,
                   As, Bs, Gs, wgB0, wgN0);
        grid_barrier(bar_cnt, bar_ph, ++phase);

        // layer 1: consumes h0[p^1] and h1[p], produces h1[p^1], c1, y[t]
        lstm_layer(h0ws + (size_t)(p ^ 1) * BHf,
                   Wih1 + (size_t)t * G4 * HH,
                   h1ws + (size_t)p * BHf,
                   Whh1 + (size_t)t * G4 * HH,
                   bih1 + (size_t)t * G4, bhh1 + (size_t)t * G4,
                   c1ws, h1ws + (size_t)(p ^ 1) * BHf,
                   out + (size_t)t * BB * HH,
                   As, Bs, Gs, wgB0, wgN0);
        grid_barrier(bar_cnt, bar_ph, ++phase);

        p ^= 1;
    }
}

extern "C" void kernel_launch(void* const* d_in, const int* in_sizes, int n_in,
                              void* d_out, int out_size, void* d_ws, size_t ws_size,
                              hipStream_t stream) {
    (void)in_sizes; (void)n_in; (void)out_size; (void)ws_size;
    const float* x    = (const float*)d_in[0];
    const float* Wih0 = (const float*)d_in[1];
    const float* Whh0 = (const float*)d_in[2];
    const float* bih0 = (const float*)d_in[3];
    const float* bhh0 = (const float*)d_in[4];
    const float* Wih1 = (const float*)d_in[5];
    const float* Whh1 = (const float*)d_in[6];
    const float* bih1 = (const float*)d_in[7];
    const float* bhh1 = (const float*)d_in[8];
    float* out = (float*)d_out;
    float* ws  = (float*)d_ws;

    // reset barrier state + h/c state every call (graph-replay safe)
    lstm70_init_ws<<<768, 256, 0, stream>>>(ws);
    lstm70_persistent<<<NWG, 256, 0, stream>>>(
        x, Wih0, Whh0, bih0, bhh0, Wih1, Whh1, bih1, bhh1, out, ws);
}
